// FedGNN_75247827026509
// MI455X (gfx1250) — matrix operand order repaired and checked by reading, not compiled, not run on hardware
//
#include <hip/hip_runtime.h>
#include <math.h>

#define N_NODES 50000
#define E_EDGES 800000
#define E_TOT   (E_EDGES + N_NODES)   /* 850000 with self loops */
#define F_IN    64
#define HID     64
#define NCLS    10
#define NGRAPH  512
#define NEG_SLOPE 0.2f
#define ROWTILES (N_NODES / 16)       /* 3125, exact */

typedef __attribute__((ext_vector_type(16))) _Float16 v16h;
typedef __attribute__((ext_vector_type(8)))  _Float16 v8h;
typedef __attribute__((ext_vector_type(8)))  float    v8f;

// ---- ordered-int encoding so float max can use integer atomicMax -----------
__device__ __forceinline__ int f2ord(float f) {
  int i = __float_as_int(f);
  return i >= 0 ? i : (i ^ 0x7fffffff);
}
__device__ __forceinline__ float ord2f(int i) {
  return __int_as_float(i >= 0 ? i : (i ^ 0x7fffffff));
}
#define ORD_NEG_INF ((int)0x807fffff)   /* f2ord(-inf) */

// ---- async global->LDS staging (CDNA5 ASYNCcnt path) -----------------------
// One contiguous panel of n16 16-byte transfers. Generic shared pointer's low
// 32 bits are the LDS byte offset.
__device__ __forceinline__ void stage_panel_async(float* __restrict__ ldst,
                                                  const float* __restrict__ gsrc,
                                                  int n16, int tid, int nthr) {
  for (int i = tid; i < n16; i += nthr) {
    unsigned loff = (unsigned)(size_t)(const void*)(ldst + i * 4);
    unsigned long long ga = (unsigned long long)(size_t)(gsrc + i * 4);
    asm volatile("global_load_async_to_lds_b128 %0, %1, off"
                 :: "v"(loff), "v"(ga) : "memory");
  }
}
__device__ __forceinline__ void wait_async0() {
  asm volatile("s_wait_asynccnt 0x0" ::: "memory");
}

// ---- dual GEMM: OUTL = X@WL, OUTR = X@WR via v_wmma_f32_16x16x32_f16 -------
// X: [M,K] fp32 row-major, W: [K,NC] fp32 row-major, OUT: [M,NC] fp32.
// Block = 2*(NC/16) waves; each wave owns one 16x16 tile of the L or R product.
// Weights staged once per block into LDS transposed as f16 (sW[n*K+k]);
// A panels (16 contiguous rows) double-buffered via async loads to LDS.
template <int K, int NC, int RPB>
__global__ __launch_bounds__(2 * (NC / 16) * 32)
void wmma_dual_gemm(const float* __restrict__ X,
                    const float* __restrict__ WL,
                    const float* __restrict__ WR,
                    float* __restrict__ OUTL,
                    float* __restrict__ OUTR) {
  constexpr int NTILES = NC / 16;
  constexpr int BD     = 2 * NTILES * 32;   // threads per block
  constexpr int NT16   = 16 * K / 4;        // 16B transfers per A panel

  __shared__ _Float16 sW[2 * NC * K];       // [mat][n][k], transposed f16
  __shared__ float    sA[2][16 * K];        // double-buffered fp32 A panel

  const int tid  = threadIdx.x;
  const int lane = tid & 31;
  const int wave = tid >> 5;
  const int mat  = wave / NTILES;           // 0 -> L, 1 -> R
  const int ct   = wave % NTILES;           // column tile
  const int l16  = lane & 15;
  const int n    = (ct << 4) + l16;         // B/D column
  const int kA   = (lane < 16) ? 0 : 8;     // 16-bit A layout K base
  const int kB   = (lane < 16) ? 0 : 8;     // 16-bit B layout K base

  const float* WSRC0 = WL;
  const float* WSRC1 = WR;
  // stage both weight matrices into LDS, transposed, converted to f16
  for (int i = tid; i < K * NC; i += BD) {
    int k = i / NC, c = i - k * NC;
    sW[c * K + k]          = (_Float16)WSRC0[i];
    sW[NC * K + c * K + k] = (_Float16)WSRC1[i];
  }

  const int rt0 = blockIdx.x * RPB;
  int nrt = ROWTILES - rt0;                 // block-uniform tail
  if (nrt > RPB) nrt = RPB;

  // prologue: async-stage first A panel into buffer 0
  stage_panel_async(sA[0], X + (size_t)rt0 * 16 * K, NT16, tid, BD);
  wait_async0();
  __syncthreads();                          // weights + panel 0 visible

  const _Float16* Wt = sW + (size_t)mat * NC * K + (size_t)n * K;
  int buf = 0;
  for (int i = 0; i < nrt; ++i) {
    if (i + 1 < nrt)                        // prefetch next panel (overlapped)
      stage_panel_async(sA[buf ^ 1], X + (size_t)(rt0 + i + 1) * 16 * K,
                        NT16, tid, BD);

    const float* Ar = sA[buf] + l16 * K;    // this lane's A row
    v8f acc = {};
#pragma unroll
    for (int kk = 0; kk < K; kk += 32) {
      // A tile 16x32: e<8 -> k=kA+e, e>=8 -> k=16+kA+(e-8)  (2x32B ds reads)
      const float4* p = (const float4*)(Ar + kk + kA);
      const float4* q = (const float4*)(Ar + kk + 16 + kA);
      float4 f0 = p[0], f1 = p[1], f2 = q[0], f3 = q[1];
      v16h a;
      a[0]  = (_Float16)f0.x; a[1]  = (_Float16)f0.y;
      a[2]  = (_Float16)f0.z; a[3]  = (_Float16)f0.w;
      a[4]  = (_Float16)f1.x; a[5]  = (_Float16)f1.y;
      a[6]  = (_Float16)f1.z; a[7]  = (_Float16)f1.w;
      a[8]  = (_Float16)f2.x; a[9]  = (_Float16)f2.y;
      a[10] = (_Float16)f2.z; a[11] = (_Float16)f2.w;
      a[12] = (_Float16)f3.x; a[13] = (_Float16)f3.y;
      a[14] = (_Float16)f3.z; a[15] = (_Float16)f3.w;
      // B tile 32x16: e<8 -> k=kB+e, e>=8 -> k=16+kB+(e-8)
      v8h b0 = *(const v8h*)(Wt + kk + kB);
      v8h b1 = *(const v8h*)(Wt + kk + 16 + kB);
      v16h b = __builtin_shufflevector(b0, b1, 0, 1, 2, 3, 4, 5, 6, 7,
                                       8, 9, 10, 11, 12, 13, 14, 15);
      acc = __builtin_amdgcn_wmma_f32_16x16x32_f16(
          false, a, false, b, (short)0, acc, false, false);
    }

    // D: VGPR r -> row r (lanes 0-15) / 8+r (lanes 16-31); coalesced over n
    float* OUT = mat ? OUTR : OUTL;
    const int rowbase = (rt0 + i) << 4;
#pragma unroll
    for (int r = 0; r < 8; ++r) {
      int mr = rowbase + ((lane < 16) ? r : 8 + r);
      OUT[(size_t)mr * NC + n] = acc[r];
    }

    wait_async0();                          // next panel landed
    __syncthreads();                        // all reads of buf done, publish
    buf ^= 1;
  }
}

// ---- trivial fills ---------------------------------------------------------
__global__ void fill_f32(float* __restrict__ p, size_t n, float v) {
  size_t i = (size_t)blockIdx.x * blockDim.x + threadIdx.x;
  if (i < n) p[i] = v;
}
__global__ void fill_i32(int* __restrict__ p, size_t n, int v) {
  size_t i = (size_t)blockIdx.x * blockDim.x + threadIdx.x;
  if (i < n) p[i] = v;
}

// ---- edge indexing with implicit self loops --------------------------------
__device__ __forceinline__ void edge_sd(const int* __restrict__ ei, int e,
                                        int& s, int& d) {
  if (e < E_EDGES) { s = ei[e]; d = ei[E_EDGES + e]; }
  else             { s = e - E_EDGES; d = s; }
}

// pass 1: logits = att_h . leaky_relu(xl[src]+xr[dst]); segment max into M
__global__ void edge_logits(const int* __restrict__ ei,
                            const float* __restrict__ XL,
                            const float* __restrict__ XR,
                            const float* __restrict__ att,
                            float* __restrict__ LOG,
                            int* __restrict__ M, int H) {
  int t = blockIdx.x * blockDim.x + threadIdx.x;
  if (t >= E_TOT * H) return;
  int e = t / H, h = t - e * H;
  int s, d; edge_sd(ei, e, s, d);
  const float4* xl = (const float4*)(XL + ((size_t)s * H + h) * HID);
  const float4* xr = (const float4*)(XR + ((size_t)d * H + h) * HID);
  const float4* a  = (const float4*)(att + h * HID);
  float acc = 0.f;
#pragma unroll 4
  for (int c = 0; c < HID / 4; ++c) {
    float4 l = xl[c], r = xr[c], w = a[c];
    float v0 = l.x + r.x, v1 = l.y + r.y, v2 = l.z + r.z, v3 = l.w + r.w;
    v0 = v0 > 0.f ? v0 : NEG_SLOPE * v0;
    v1 = v1 > 0.f ? v1 : NEG_SLOPE * v1;
    v2 = v2 > 0.f ? v2 : NEG_SLOPE * v2;
    v3 = v3 > 0.f ? v3 : NEG_SLOPE * v3;
    acc = fmaf(w.x, v0, fmaf(w.y, v1, fmaf(w.z, v2, fmaf(w.w, v3, acc))));
  }
  LOG[t] = acc;
  atomicMax(&M[(size_t)d * H + h], f2ord(acc));
}

// pass 2: ex = exp(logit - max[dst]); segment sum into DEN
__global__ void edge_exp(const int* __restrict__ ei,
                         float* __restrict__ LOG,
                         const int* __restrict__ M,
                         float* __restrict__ DEN, int H) {
  int t = blockIdx.x * blockDim.x + threadIdx.x;
  if (t >= E_TOT * H) return;
  int e = t / H, h = t - e * H;
  int s, d; edge_sd(ei, e, s, d);
  (void)s;
  float ex = __expf(LOG[t] - ord2f(M[(size_t)d * H + h]));
  LOG[t] = ex;
  atomicAdd(&DEN[(size_t)d * H + h], ex);
}

// pass 3: out[dst] += (ex/denom[dst]) * xl[src]
__global__ void edge_aggregate(const int* __restrict__ ei,
                               const float* __restrict__ LOG,
                               const float* __restrict__ DEN,
                               const float* __restrict__ XL,
                               float* __restrict__ AGG, int H) {
  int t = blockIdx.x * blockDim.x + threadIdx.x;
  if (t >= E_TOT * H) return;
  int e = t / H, h = t - e * H;
  int s, d; edge_sd(ei, e, s, d);
  float alpha = LOG[t] / DEN[(size_t)d * H + h];
  const float4* xl = (const float4*)(XL + ((size_t)s * H + h) * HID);
  float* o         = AGG + ((size_t)d * H + h) * HID;
#pragma unroll 4
  for (int c = 0; c < HID / 4; ++c) {
    float4 v = xl[c];
    atomicAdd(&o[4 * c + 0], alpha * v.x);
    atomicAdd(&o[4 * c + 1], alpha * v.y);
    atomicAdd(&o[4 * c + 2], alpha * v.z);
    atomicAdd(&o[4 * c + 3], alpha * v.w);
  }
}

// bias (+ optional ELU) applied in place
__global__ void bias_act(float* __restrict__ A, const float* __restrict__ b,
                         int NC, int do_elu, size_t total) {
  size_t i = (size_t)blockIdx.x * blockDim.x + threadIdx.x;
  if (i >= total) return;
  int j = (int)(i % NC);
  float v = A[i] + b[j];
  if (do_elu) v = v > 0.f ? v : (__expf(v) - 1.f);
  A[i] = v;
}

// global mean pool: accumulate sums + counts per graph
__global__ void pool_accum(const float* __restrict__ H2,
                           const int* __restrict__ batch,
                           float* __restrict__ SUMS, float* __restrict__ CNT) {
  int t = blockIdx.x * blockDim.x + threadIdx.x;
  if (t >= N_NODES * HID) return;
  int nid = t / HID, c = t - nid * HID;
  int g = batch[nid];
  atomicAdd(&SUMS[(size_t)g * HID + c], H2[t]);
  if (c == 0) atomicAdd(&CNT[g], 1.0f);
}

// final classifier: out = (sums/cnt) @ lin_w + lin_b
__global__ void classify(const float* __restrict__ SUMS,
                         const float* __restrict__ CNT,
                         const float* __restrict__ lw,
                         const float* __restrict__ lb,
                         float* __restrict__ out) {
  int t = blockIdx.x * blockDim.x + threadIdx.x;
  if (t >= NGRAPH * NCLS) return;
  int g = t / NCLS, k = t - g * NCLS;
  float inv = 1.0f / fmaxf(CNT[g], 1.0f);
  float acc = lb[k];
#pragma unroll 8
  for (int c = 0; c < HID; ++c)
    acc = fmaf(SUMS[(size_t)g * HID + c] * inv, lw[c * NCLS + k], acc);
  out[t] = acc;
}

static inline int cdiv(size_t a, int b) { return (int)((a + b - 1) / b); }

extern "C" void kernel_launch(void* const* d_in, const int* in_sizes, int n_in,
                              void* d_out, int out_size, void* d_ws, size_t ws_size,
                              hipStream_t stream) {
  (void)in_sizes; (void)n_in; (void)out_size; (void)ws_size;
  const float* x    = (const float*)d_in[0];
  const int*   ei   = (const int*)d_in[1];
  const int*   batch= (const int*)d_in[2];
  const float* Wl1  = (const float*)d_in[3];
  const float* Wr1  = (const float*)d_in[4];
  const float* att1 = (const float*)d_in[5];
  const float* b1   = (const float*)d_in[6];
  const float* Wl2  = (const float*)d_in[7];
  const float* Wr2  = (const float*)d_in[8];
  const float* att2 = (const float*)d_in[9];
  const float* b2   = (const float*)d_in[10];
  const float* lw   = (const float*)d_in[11];
  const float* lb   = (const float*)d_in[12];
  float* out = (float*)d_out;

  // workspace layout (floats)
  float* ws = (float*)d_ws;
  size_t off = 0;
  float* XL1 = ws + off; off += (size_t)N_NODES * 128;
  float* XR1 = ws + off; off += (size_t)N_NODES * 128;
  float* H1  = ws + off; off += (size_t)N_NODES * 128;   // agg1 -> h1 in place
  float* XL2 = ws + off; off += (size_t)N_NODES * 64;
  float* XR2 = ws + off; off += (size_t)N_NODES * 64;
  float* H2  = ws + off; off += (size_t)N_NODES * 64;    // agg2 -> h2 in place
  float* LOG = ws + off; off += (size_t)E_TOT * 2;       // reused by conv2 (H=1)
  float* DEN = ws + off; off += (size_t)N_NODES * 2;
  int*   M   = (int*)(ws + off); off += (size_t)N_NODES * 2;
  float* SUMS= ws + off; off += (size_t)NGRAPH * HID;
  float* CNT = ws + off; off += (size_t)NGRAPH;

  const int TB  = 256;
  const int RPB = 8;
  const int GGRID = cdiv(ROWTILES, RPB);   // 391 blocks, uniform tail

  // ---------------- conv1: GATv2(64 -> 64, heads=2) ----------------
  wmma_dual_gemm<64, 128, 8><<<GGRID, 512, 0, stream>>>(x, Wl1, Wr1, XL1, XR1);

  fill_i32<<<cdiv((size_t)N_NODES * 2, TB), TB, 0, stream>>>(M, (size_t)N_NODES * 2, ORD_NEG_INF);
  fill_f32<<<cdiv((size_t)N_NODES * 2, TB), TB, 0, stream>>>(DEN, (size_t)N_NODES * 2, 0.f);
  fill_f32<<<cdiv((size_t)N_NODES * 128, TB), TB, 0, stream>>>(H1, (size_t)N_NODES * 128, 0.f);

  edge_logits   <<<cdiv((size_t)E_TOT * 2, TB), TB, 0, stream>>>(ei, XL1, XR1, att1, LOG, M, 2);
  edge_exp      <<<cdiv((size_t)E_TOT * 2, TB), TB, 0, stream>>>(ei, LOG, M, DEN, 2);
  edge_aggregate<<<cdiv((size_t)E_TOT * 2, TB), TB, 0, stream>>>(ei, LOG, DEN, XL1, H1, 2);
  bias_act      <<<cdiv((size_t)N_NODES * 128, TB), TB, 0, stream>>>(H1, b1, 128, 1, (size_t)N_NODES * 128);

  // ---------------- conv2: GATv2(128 -> 64, heads=1) ----------------
  wmma_dual_gemm<128, 64, 8><<<GGRID, 256, 0, stream>>>(H1, Wl2, Wr2, XL2, XR2);

  fill_i32<<<cdiv((size_t)N_NODES, TB), TB, 0, stream>>>(M, (size_t)N_NODES, ORD_NEG_INF);
  fill_f32<<<cdiv((size_t)N_NODES, TB), TB, 0, stream>>>(DEN, (size_t)N_NODES, 0.f);
  fill_f32<<<cdiv((size_t)N_NODES * 64, TB), TB, 0, stream>>>(H2, (size_t)N_NODES * 64, 0.f);

  edge_logits   <<<cdiv((size_t)E_TOT, TB), TB, 0, stream>>>(ei, XL2, XR2, att2, LOG, M, 1);
  edge_exp      <<<cdiv((size_t)E_TOT, TB), TB, 0, stream>>>(ei, LOG, M, DEN, 1);
  edge_aggregate<<<cdiv((size_t)E_TOT, TB), TB, 0, stream>>>(ei, LOG, DEN, XL2, H2, 1);
  bias_act      <<<cdiv((size_t)N_NODES * 64, TB), TB, 0, stream>>>(H2, b2, 64, 0, (size_t)N_NODES * 64);

  // ---------------- mean pool + classifier ----------------
  fill_f32<<<cdiv((size_t)NGRAPH * HID, TB), TB, 0, stream>>>(SUMS, (size_t)NGRAPH * HID, 0.f);
  fill_f32<<<cdiv((size_t)NGRAPH, TB), TB, 0, stream>>>(CNT, (size_t)NGRAPH, 0.f);
  pool_accum<<<cdiv((size_t)N_NODES * HID, TB), TB, 0, stream>>>(H2, batch, SUMS, CNT);
  classify  <<<cdiv((size_t)NGRAPH * NCLS, TB), TB, 0, stream>>>(SUMS, CNT, lw, lb, out);
}
